// MultiHeadAttention_77515569758310
// MI455X (gfx1250) — matrix operand order repaired and checked by reading, not compiled, not run on hardware
//
#include <hip/hip_runtime.h>

// ---------------------------------------------------------------------------
// MI455X (gfx1250) multi-head attention, bf16 WMMA pipeline.
//   EMB=256, HEADS=8, head_dim=32, B=4, N=2048  (rows R = B*N = 8192)
// ---------------------------------------------------------------------------

#define EMB   256
#define HEADS 8
#define HD    32          // head dim == WMMA K for bf16 (16x16x32)
#define BATCH 4
#define SEQ   2048
#define ROWS  (BATCH * SEQ)   // 8192

typedef __bf16 v16bf __attribute__((ext_vector_type(16)));
typedef __bf16 v8bf  __attribute__((ext_vector_type(8)));
typedef __bf16 v4bf  __attribute__((ext_vector_type(4)));
typedef float  v8f   __attribute__((ext_vector_type(8)));

__device__ __forceinline__ v4bf f4_to_bf(float4 v) {
    v4bf r;
    r[0] = (__bf16)v.x; r[1] = (__bf16)v.y;
    r[2] = (__bf16)v.z; r[3] = (__bf16)v.w;
    return r;   // native v_cvt_pk_bf16_f32 path
}

__device__ __forceinline__ v16bf cat8(v8bf lo, v8bf hi) {
    return __builtin_shufflevector(lo, hi, 0,1,2,3,4,5,6,7,8,9,10,11,12,13,14,15);
}

__device__ __forceinline__ v8f wmma_bf16(v16bf a, v16bf b, v8f c) {
    // D = A(16x32 bf16) * B(32x16 bf16) + C(16x16 f32)
    return __builtin_amdgcn_wmma_f32_16x16x32_bf16(
        /*neg_a=*/false, a, /*neg_b=*/false, b,
        /*c_mod=*/(short)0, c, /*reuse_a=*/false, /*reuse_b=*/false);
}

// ---------------------------------------------------------------------------
// Kernel 1: fused QKV projection  Y = x @ W^T + b  (blockIdx.z selects q/k/v)
// Output stored bf16 in (b, h, n, d) layout for the attention kernel.
// Q is pre-scaled by 1/sqrt(EMB) = 1/16.
// Block: 256 thr (8 waves). Tile: M=64, N=64, Kstep=32. Wave (wm,wn) -> 16x32.
// ---------------------------------------------------------------------------
__global__ void __launch_bounds__(256)
qkv_kernel(const float* __restrict__ x,
           const float* __restrict__ Wq, const float* __restrict__ bq,
           const float* __restrict__ Wk, const float* __restrict__ bk,
           const float* __restrict__ Wv, const float* __restrict__ bv,
           __bf16* __restrict__ Q, __bf16* __restrict__ K,
           __bf16* __restrict__ V)
{
    __shared__ __align__(16) __bf16 Xs[64 * 32];
    __shared__ __align__(16) __bf16 Ws[64 * 32];

    const int z = blockIdx.z;
    const float* W    = (z == 0) ? Wq : (z == 1) ? Wk : Wv;
    const float* bias = (z == 0) ? bq : (z == 1) ? bk : bv;
    __bf16* dst = (z == 0) ? Q : (z == 1) ? K : V;

    const int r0 = blockIdx.x * 64;   // row tile in x (b*N+n space)
    const int o0 = blockIdx.y * 64;   // output-feature tile
    const int t    = threadIdx.x;
    const int lane = t & 31, w = t >> 5;
    const int wm = w >> 1, wn = w & 1;
    const int ln = lane & 15, hi = lane >> 4;

    v8f acc0 = {}; v8f acc1 = {};

    for (int k0 = 0; k0 < EMB; k0 += 32) {
        // -------- stage: X(64x32) and W(64x32) as bf16 in LDS --------
        #pragma unroll
        for (int i = 0; i < 2; ++i) {
            const int idx = t + i * 256;           // 512 float4 chunks
            const int row = idx >> 3;
            const int c4  = (idx & 7) * 4;
            float4 xv = *(const float4*)(x + (size_t)(r0 + row) * EMB + k0 + c4);
            *(v4bf*)(&Xs[row * 32 + c4]) = f4_to_bf(xv);
            float4 wv = *(const float4*)(W + (size_t)(o0 + row) * EMB + k0 + c4);
            *(v4bf*)(&Ws[row * 32 + c4]) = f4_to_bf(wv);
        }
        if (k0 + 32 < EMB) {  // pull next tiles toward the WGP (global_prefetch_b8)
            __builtin_prefetch(x + (size_t)(r0 + (t >> 2)) * EMB + k0 + 32, 0, 3);
            __builtin_prefetch(W + (size_t)(o0 + (t >> 2)) * EMB + k0 + 32, 0, 3);
        }
        __syncthreads();

        // -------- fragments (CDNA5 wave32 layouts) --------
        // A 16x32: lane holds row=ln; k = hi*16 + e contiguous
        const int arow = wm * 16 + ln;
        v16bf afrag = cat8(*(const v8bf*)(&Xs[arow * 32 + hi * 16]),
                           *(const v8bf*)(&Xs[arow * 32 + hi * 16 + 8]));
        // B 32x16: lane holds col=ln (output feature); k = hi*16 + e contiguous
        v16bf b0 = *(const v16bf*)(&Ws[(wn * 32 +      ln) * 32 + hi * 16]);
        v16bf b1 = *(const v16bf*)(&Ws[(wn * 32 + 16 + ln) * 32 + hi * 16]);

        acc0 = wmma_bf16(afrag, b0, acc0);
        acc1 = wmma_bf16(afrag, b1, acc1);
        __syncthreads();
    }

    // -------- epilogue: bias, (Q-only) 1/sqrt(E) scale, bf16 store --------
    const float sc = (z == 0) ? 0.0625f : 1.0f;
    #pragma unroll
    for (int r = 0; r < 8; ++r) {
        const int m   = wm * 16 + hi * 8 + r;     // C layout: row = hi*8 + r
        const int row = r0 + m;
        const int b   = row >> 11;                 // / SEQ
        const int n   = row & (SEQ - 1);
        #pragma unroll
        for (int g = 0; g < 2; ++g) {
            const int o = o0 + wn * 32 + g * 16 + ln;
            const float val = ((g ? acc1[r] : acc0[r]) + bias[o]) * sc;
            const int h = o >> 5, dd = o & 31;
            dst[(((size_t)b * HEADS + h) * SEQ + n) * HD + dd] = (__bf16)val;
        }
    }
}

// ---------------------------------------------------------------------------
// Kernel 2: flash attention per (b,h). Block = 8 waves, each wave owns a
// 16-row Q tile (128 q rows / block). Streams K/V in 32-key tiles via LDS.
// ---------------------------------------------------------------------------
__global__ void __launch_bounds__(256)
attn_kernel(const __bf16* __restrict__ Q,
            const __bf16* __restrict__ K,
            const __bf16* __restrict__ V,
            __bf16* __restrict__ O)    // (b, n, h*d) row-major bf16
{
    __shared__ __align__(16) __bf16 Ks[32 * 32];        // [key][d]
    __shared__ __align__(16) __bf16 Vt[32 * 32];        // [d][key] (transposed)
    __shared__ __align__(16) __bf16 Ps[8 * 16 * 32];    // per-wave P tile

    const int bh = blockIdx.y;                // b*HEADS + h
    const int b  = bh >> 3, h = bh & 7;
    const int q0 = blockIdx.x * 128;
    const int t = threadIdx.x, lane = t & 31, w = t >> 5;
    const int ln = lane & 15, hi = lane >> 4;
    const size_t base = (size_t)bh * SEQ * HD;

    // Q fragment: A 16x32 (k = head dim = 32), resident all kernel.
    const __bf16* qp = Q + base + (size_t)(q0 + w * 16 + ln) * HD;
    const v16bf qfrag = cat8(*(const v8bf*)(qp + hi * 16),
                             *(const v8bf*)(qp + hi * 16 + 8));

    v8f o0 = {}; v8f o1 = {};
    float mrow[8], lrow[8];
    #pragma unroll
    for (int r = 0; r < 8; ++r) { mrow[r] = -3.0e38f; lrow[r] = 0.0f; }

    __bf16* pw = &Ps[w * 16 * 32];

    for (int kt = 0; kt < SEQ; kt += 32) {
        // -------- cooperative stage of K (row-major) and V (transposed) ----
        if (t < 128) {
            const int key = t >> 2, c = (t & 3) * 8;
            v8bf kv = *(const v8bf*)(K + base + (size_t)(kt + key) * HD + c);
            *(v8bf*)(&Ks[key * 32 + c]) = kv;
        } else {
            const int t2 = t - 128;
            const int key = t2 >> 2, c = (t2 & 3) * 8;
            v8bf vv = *(const v8bf*)(V + base + (size_t)(kt + key) * HD + c);
            #pragma unroll
            for (int i = 0; i < 8; ++i)
                Vt[(c + i) * 32 + key] = vv[i];
        }
        __syncthreads();

        // -------- S = Q @ K^T : two 16-key groups, contraction K = d = 32 --
        v16bf kb0 = *(const v16bf*)(&Ks[(     ln) * 32 + hi * 16]);
        v16bf kb1 = *(const v16bf*)(&Ks[(16 + ln) * 32 + hi * 16]);
        const v8f zc = {};
        v8f s0 = wmma_bf16(qfrag, kb0, zc);
        v8f s1 = wmma_bf16(qfrag, kb1, zc);

        // -------- fp32 online softmax (rows = hi*8 + r, cols across 16 lanes)
        float p0a[8], p1a[8];
        #pragma unroll
        for (int r = 0; r < 8; ++r) {
            float s0r = s0[r], s1r = s1[r];
            float mx = fmaxf(s0r, s1r);
            #pragma unroll
            for (int msk = 8; msk > 0; msk >>= 1)
                mx = fmaxf(mx, __shfl_xor(mx, msk, 32));   // stays in 16-lane half
            const float mnew  = fmaxf(mrow[r], mx);
            const float alpha = __expf(mrow[r] - mnew);
            const float e0 = __expf(s0r - mnew);
            const float e1 = __expf(s1r - mnew);
            float ps = e0 + e1;
            #pragma unroll
            for (int msk = 8; msk > 0; msk >>= 1)
                ps += __shfl_xor(ps, msk, 32);
            lrow[r] = lrow[r] * alpha + ps;
            mrow[r] = mnew;
            o0[r] *= alpha; o1[r] *= alpha;    // rescale accumulator (passed as C)
            p0a[r] = e0; p1a[r] = e1;
        }

        // -------- P (C-layout) -> LDS -> A fragment; same-wave DS is in-order
        #pragma unroll
        for (int r = 0; r < 8; ++r) {
            const int ml = hi * 8 + r;
            pw[ml * 32 +      ln] = (__bf16)p0a[r];
            pw[ml * 32 + 16 + ln] = (__bf16)p1a[r];
        }
        asm volatile("s_wait_dscnt 0" ::: "memory");
        v16bf pfrag = cat8(*(const v8bf*)(&pw[ln * 32 + hi * 16]),
                           *(const v8bf*)(&pw[ln * 32 + hi * 16 + 8]));

        // -------- O += P @ V : B fragments from transposed V tile ----------
        v16bf vb0 = *(const v16bf*)(&Vt[(     ln) * 32 + hi * 16]);
        v16bf vb1 = *(const v16bf*)(&Vt[(16 + ln) * 32 + hi * 16]);
        o0 = wmma_bf16(pfrag, vb0, o0);
        o1 = wmma_bf16(pfrag, vb1, o1);
        __syncthreads();
    }

    // -------- normalize and store (b, n, h*d) bf16 -------------------------
    #pragma unroll
    for (int r = 0; r < 8; ++r) {
        const float inv = 1.0f / lrow[r];
        const int n = q0 + w * 16 + hi * 8 + r;
        const size_t rowBN = (size_t)b * SEQ + n;
        O[rowBN * EMB + h * HD +      ln] = (__bf16)(o0[r] * inv);
        O[rowBN * EMB + h * HD + 16 + ln] = (__bf16)(o1[r] * inv);
    }
}

// ---------------------------------------------------------------------------
// Kernel 3: output projection  out = attn @ Wo^T + bo  (f32 result)
// ---------------------------------------------------------------------------
__global__ void __launch_bounds__(256)
out_proj_kernel(const __bf16* __restrict__ A,   // 8192 x 256 bf16
                const float* __restrict__ Wo, const float* __restrict__ bo,
                float* __restrict__ out)
{
    __shared__ __align__(16) __bf16 Xs[64 * 32];
    __shared__ __align__(16) __bf16 Ws[64 * 32];

    const int r0 = blockIdx.x * 64;
    const int o0 = blockIdx.y * 64;
    const int t = threadIdx.x, lane = t & 31, w = t >> 5;
    const int wm = w >> 1, wn = w & 1;
    const int ln = lane & 15, hi = lane >> 4;

    v8f acc0 = {}; v8f acc1 = {};

    for (int k0 = 0; k0 < EMB; k0 += 32) {
        {   // A tile: already bf16 -> straight 16B copies
            const int row = t >> 2, c8 = (t & 3) * 8;
            v8bf av = *(const v8bf*)(A + (size_t)(r0 + row) * EMB + k0 + c8);
            *(v8bf*)(&Xs[row * 32 + c8]) = av;
        }
        #pragma unroll
        for (int i = 0; i < 2; ++i) {  // Wo tile: f32 -> bf16
            const int idx = t + i * 256;
            const int row = idx >> 3, c4 = (idx & 7) * 4;
            float4 wv = *(const float4*)(Wo + (size_t)(o0 + row) * EMB + k0 + c4);
            *(v4bf*)(&Ws[row * 32 + c4]) = f4_to_bf(wv);
        }
        if (k0 + 32 < EMB)
            __builtin_prefetch(Wo + (size_t)(o0 + (t >> 2)) * EMB + k0 + 32, 0, 3);
        __syncthreads();

        const int arow = wm * 16 + ln;
        v16bf afrag = cat8(*(const v8bf*)(&Xs[arow * 32 + hi * 16]),
                           *(const v8bf*)(&Xs[arow * 32 + hi * 16 + 8]));
        v16bf b0 = *(const v16bf*)(&Ws[(wn * 32 +      ln) * 32 + hi * 16]);
        v16bf b1 = *(const v16bf*)(&Ws[(wn * 32 + 16 + ln) * 32 + hi * 16]);
        acc0 = wmma_bf16(afrag, b0, acc0);
        acc1 = wmma_bf16(afrag, b1, acc1);
        __syncthreads();
    }

    #pragma unroll
    for (int r = 0; r < 8; ++r) {
        const int row = r0 + wm * 16 + hi * 8 + r;
        #pragma unroll
        for (int g = 0; g < 2; ++g) {
            const int o = o0 + wn * 32 + g * 16 + ln;
            out[(size_t)row * EMB + o] = (g ? acc1[r] : acc0[r]) + bo[o];
        }
    }
}

// ---------------------------------------------------------------------------
extern "C" void kernel_launch(void* const* d_in, const int* in_sizes, int n_in,
                              void* d_out, int out_size, void* d_ws, size_t ws_size,
                              hipStream_t stream)
{
    const float* x  = (const float*)d_in[0];
    const float* Wq = (const float*)d_in[1];
    const float* bq = (const float*)d_in[2];
    const float* Wk = (const float*)d_in[3];
    const float* bk = (const float*)d_in[4];
    const float* Wv = (const float*)d_in[5];
    const float* bv = (const float*)d_in[6];
    const float* Wo = (const float*)d_in[7];
    const float* bo = (const float*)d_in[8];

    // Workspace: Q | K | V | attn_out, all bf16 (16 MiB total)
    const size_t perQKV = (size_t)BATCH * HEADS * SEQ * HD;   // 2,097,152 elems
    __bf16* Qw = (__bf16*)d_ws;
    __bf16* Kw = Qw + perQKV;
    __bf16* Vw = Kw + perQKV;
    __bf16* AT = Vw + perQKV;                          // 8192 x 256

    dim3 blk(256);
    qkv_kernel<<<dim3(ROWS / 64, EMB / 64, 3), blk, 0, stream>>>(
        x, Wq, bq, Wk, bk, Wv, bv, Qw, Kw, Vw);
    attn_kernel<<<dim3(SEQ / 128, BATCH * HEADS), blk, 0, stream>>>(
        Qw, Kw, Vw, AT);
    out_proj_kernel<<<dim3(ROWS / 64, EMB / 64), blk, 0, stream>>>(
        AT, Wo, bo, (float*)d_out);
}
